// RelativeMultiHeadAttention_9156870275493
// MI455X (gfx1250) — matrix operand order repaired and checked by reading, not compiled, not run on hardware
//
#include <hip/hip_runtime.h>
#include <hip/hip_bf16.h>

#define TDIM 1024
#define BB 4
#define NH 16
#define HD 64
#define ED 1024
#define NTOK (BB * TDIM)   // 4096 tokens
#define RELN 1999          // 2*MAX_LEN - 1
#define SW 48              // rel-pos P window width (32 + 15, padded to 48)

typedef __attribute__((ext_vector_type(16))) __bf16 v16bf;
typedef __attribute__((ext_vector_type(8)))  float  v8f;

union FragBF {
    v16bf v;
    unsigned int u[8];
};

__device__ inline v8f zero8() {
    v8f z = {0.f, 0.f, 0.f, 0.f, 0.f, 0.f, 0.f, 0.f};
    return z;
}

__device__ inline v8f wmma_bf16(v16bf a, v16bf b, v8f c) {
    // v_wmma_f32_16x16x32_bf16: D = A(16x32) x B(32x16) + C(16x16 f32)
    return __builtin_amdgcn_wmma_f32_16x16x32_bf16(
        /*neg_a=*/false, a, /*neg_b=*/false, b,
        /*c_mod=*/(short)0, c, /*reuse_a=*/false, /*reuse_b=*/false);
}

// A-fragment: halfs 0..7 at base[0..7], halfs 8..15 at base[16..23].
// Caller passes base = row_ptr + kstep + hi*8 (ISA 16-bit A layout, wave32).
__device__ inline v16bf load_frag_A(const __bf16* base) {
    FragBF f;
    const uint4* p0 = reinterpret_cast<const uint4*>(base);
    const uint4* p1 = reinterpret_cast<const uint4*>(base + 16);
    uint4 a = *p0, b = *p1;
    f.u[0] = a.x; f.u[1] = a.y; f.u[2] = a.z; f.u[3] = a.w;
    f.u[4] = b.x; f.u[5] = b.y; f.u[6] = b.z; f.u[7] = b.w;
    return f.v;
}

// B-fragment: base = col_ptr + kstep + hi*16; loads halfs at kstep+hi*8+[0..7] and kstep+16+hi*8+[0..7] (matches A).
__device__ inline v16bf load_frag_B(const __bf16* base) {
    FragBF f;
    const __bf16* bb = base - ((threadIdx.x & 31) >> 4) * 8;
    uint4 a = *reinterpret_cast<const uint4*>(bb), b = *reinterpret_cast<const uint4*>(bb + 16);
    f.u[0] = a.x; f.u[1] = a.y; f.u[2] = a.z; f.u[3] = a.w;
    f.u[4] = b.x; f.u[5] = b.y; f.u[6] = b.z; f.u[7] = b.w;
    return f.v;
}

// ---------------------------------------------------------------------------
// Kernel 1: vectorized fp32 -> bf16 conversion (n must be a multiple of 4)
// ---------------------------------------------------------------------------
__global__ __launch_bounds__(256) void cvt4_kernel(const float* __restrict__ src,
                                                   __bf16* __restrict__ dst, int n) {
    const int i4 = (blockIdx.x * blockDim.x + threadIdx.x) * 4;
    if (i4 < n) {
        const float4 f = *reinterpret_cast<const float4*>(src + i4);
        dst[i4 + 0] = (__bf16)f.x;
        dst[i4 + 1] = (__bf16)f.y;
        dst[i4 + 2] = (__bf16)f.z;
        dst[i4 + 3] = (__bf16)f.w;
    }
}

// ---------------------------------------------------------------------------
// Kernel 2: bf16 GEMM  acc = Xb @ Wb^T + bias.  One wave = 16x64 output tile,
// software-pipelined: iteration k+1's fragments are loaded before iteration
// k's 4 WMMAs issue, so loads overlap matrix-op execution.
// MODE 0: dst bf16 head-major [B,H,T,64] (Q,K)
// MODE 1: dst bf16 transposed [B,H,64,T] (V, contiguous PV B-fragments)
// MODE 2: dst f32 row-major [token,1024]  (final output projection)
// ---------------------------------------------------------------------------
template <int MODE>
__global__ __launch_bounds__(128) void gemm_bf16_kernel(const __bf16* __restrict__ Xb,
                                                        const __bf16* __restrict__ Wb,
                                                        const float* __restrict__ bias,
                                                        void* __restrict__ dstv) {
    const int wave = threadIdx.x >> 5;
    const int lane = threadIdx.x & 31;
    const int hi = lane >> 4;
    const int lrow = lane & 15;
    const int n0 = blockIdx.x * 16;                    // token tile
    const int o0 = (blockIdx.y * 4 + wave) * 64;       // 64 output cols per wave

    const __bf16* arow = Xb + (size_t)(n0 + lrow) * ED + hi * 8;
    const __bf16* br0 = Wb + (size_t)(o0 + 0  + lrow) * ED + hi * 16;  // B[k][n] = W[o][k]
    const __bf16* br1 = Wb + (size_t)(o0 + 16 + lrow) * ED + hi * 16;
    const __bf16* br2 = Wb + (size_t)(o0 + 32 + lrow) * ED + hi * 16;
    const __bf16* br3 = Wb + (size_t)(o0 + 48 + lrow) * ED + hi * 16;

    v8f acc0 = zero8(), acc1 = zero8(), acc2 = zero8(), acc3 = zero8();

    // --- software pipeline: prologue loads for k0 = 0 ---
    v16bf A  = load_frag_A(arow);
    v16bf B0 = load_frag_B(br0);
    v16bf B1 = load_frag_B(br1);
    v16bf B2 = load_frag_B(br2);
    v16bf B3 = load_frag_B(br3);

    for (int k0 = 0; k0 < ED - 32; k0 += 32) {
        // issue next iteration's loads first (hidden behind this iter's WMMAs)
        const int kn = k0 + 32;
        const v16bf An  = load_frag_A(arow + kn);
        const v16bf Bn0 = load_frag_B(br0 + kn);
        const v16bf Bn1 = load_frag_B(br1 + kn);
        const v16bf Bn2 = load_frag_B(br2 + kn);
        const v16bf Bn3 = load_frag_B(br3 + kn);

        acc0 = wmma_bf16(A, B0, acc0);
        acc1 = wmma_bf16(A, B1, acc1);
        acc2 = wmma_bf16(A, B2, acc2);
        acc3 = wmma_bf16(A, B3, acc3);

        A = An; B0 = Bn0; B1 = Bn1; B2 = Bn2; B3 = Bn3;
    }
    // --- pipeline epilogue: last K-step ---
    acc0 = wmma_bf16(A, B0, acc0);
    acc1 = wmma_bf16(A, B1, acc1);
    acc2 = wmma_bf16(A, B2, acc2);
    acc3 = wmma_bf16(A, B3, acc3);

    v8f accs[4] = {acc0, acc1, acc2, acc3};
#pragma unroll
    for (int j = 0; j < 4; ++j) {
        const int o = o0 + j * 16 + lrow;
        const float bval = bias[o];
        const int h = o >> 6, d = o & 63;
#pragma unroll
        for (int r = 0; r < 8; ++r) {
            const int token = n0 + r + 8 * hi;
            const int b = token >> 10, t = token & 1023;
            const float val = accs[j][r] + bval;
            if (MODE == 0) {
                ((__bf16*)dstv)[((size_t)(b * NH + h) * TDIM + t) * HD + d] = (__bf16)val;
            } else if (MODE == 1) {
                ((__bf16*)dstv)[((size_t)(b * NH + h) * HD + d) * TDIM + t] = (__bf16)val;
            } else {
                ((float*)dstv)[(size_t)token * ED + o] = val;
            }
        }
    }
}

// ---------------------------------------------------------------------------
// Kernel 3: flash attention with relative-position scores.
// One wave owns a 16-row tile of one (b,h); loops keys in 32-steps.
// Per step: 4 score WMMAs + 6 rel-pos P-panel WMMAs + 4 PV WMMAs.
// V-fragments are issued at the top of the loop (consumed last -> max latency
// tolerance); K-fragments batched ahead of the score WMMAs.
// ---------------------------------------------------------------------------
__global__ __launch_bounds__(128) void attn_kernel(const __bf16* __restrict__ Qb,
                                                   const __bf16* __restrict__ Kb,
                                                   const __bf16* __restrict__ VbT,
                                                   const __bf16* __restrict__ Xqb,
                                                   const __bf16* __restrict__ relb,
                                                   __bf16* __restrict__ AttnOb) {
    __shared__ float  Pbuf[4][16 * SW];   // rel-pos P panel, per wave
    __shared__ __bf16 Ppb[4][16 * 32];    // probabilities (C-layout -> A-layout relay)

    const int wave = threadIdx.x >> 5;
    const int lane = threadIdx.x & 31;
    const int hi = lane >> 4;
    const int lrow = lane & 15;
    const int bh = blockIdx.y;
    const int b = bh >> 4, h = bh & 15;
    const int t0 = (blockIdx.x * 4 + wave) * 16;

    const __bf16* Qh = Qb  + (size_t)bh * TDIM * HD;
    const __bf16* Kh = Kb  + (size_t)bh * TDIM * HD;
    const __bf16* Vh = VbT + (size_t)bh * HD * TDIM;   // [hd][T]
    float*  Ph  = &Pbuf[wave][0];
    __bf16* Pph = &Ppb[wave][0];

    // Projected-query fragments for this row tile (hd=64 -> two K=32 A fragments).
    const v16bf qA0 = load_frag_A(Qh + (size_t)(t0 + lrow) * HD + 0  + hi * 8);
    const v16bf qA1 = load_frag_A(Qh + (size_t)(t0 + lrow) * HD + 32 + hi * 8);
    // Raw-query fragments directly from bf16 [token, 1024]: head block h, 8-groups contiguous.
    const __bf16* qrow = Xqb + (size_t)(b * TDIM + t0 + lrow) * ED + h * HD;
    const v16bf rA0 = load_frag_A(qrow + 0  + hi * 8);
    const v16bf rA1 = load_frag_A(qrow + 32 + hi * 8);

    v8f oacc0 = zero8(), oacc1 = zero8(), oacc2 = zero8(), oacc3 = zero8();
    float m[8], l[8];
#pragma unroll
    for (int r = 0; r < 8; ++r) { m[r] = -1e30f; l[r] = 0.f; }

    for (int s0 = 0; s0 < TDIM; s0 += 32) {
        // ---- issue V-fragment loads first: consumed only after softmax ----
        const v16bf vB0 = load_frag_B(Vh + (size_t)(0 * 16 + lrow) * TDIM + s0 + hi * 16);
        const v16bf vB1 = load_frag_B(Vh + (size_t)(1 * 16 + lrow) * TDIM + s0 + hi * 16);
        const v16bf vB2 = load_frag_B(Vh + (size_t)(2 * 16 + lrow) * TDIM + s0 + hi * 16);
        const v16bf vB3 = load_frag_B(Vh + (size_t)(3 * 16 + lrow) * TDIM + s0 + hi * 16);

        // ---- content scores: batch 4 K-fragment loads, then 4 WMMAs ----
        const v16bf kB00 = load_frag_B(Kh + (size_t)(s0 + lrow) * HD + hi * 16);
        const v16bf kB01 = load_frag_B(Kh + (size_t)(s0 + lrow) * HD + 32 + hi * 16);
        const v16bf kB10 = load_frag_B(Kh + (size_t)(s0 + 16 + lrow) * HD + hi * 16);
        const v16bf kB11 = load_frag_B(Kh + (size_t)(s0 + 16 + lrow) * HD + 32 + hi * 16);
        v8f Sa = zero8(), Sb = zero8();
        Sa = wmma_bf16(qA0, kB00, Sa);
        Sa = wmma_bf16(qA1, kB01, Sa);
        Sb = wmma_bf16(qA0, kB10, Sb);
        Sb = wmma_bf16(qA1, kB11, Sb);

        // ---- rel-pos panel P[t, j0+c] = qraw[t] . rel_emb[clamp(j0+c)] ----
        const int j0 = s0 - t0 - 15 + 999;   // MAX_LEN-1 = 999
#pragma unroll
        for (int nt = 0; nt < 3; ++nt) {
            int j = j0 + nt * 16 + lrow;
            j = min(max(j, 0), RELN - 1);
            const v16bf eB0 = load_frag_B(relb + (size_t)j * HD + hi * 16);
            const v16bf eB1 = load_frag_B(relb + (size_t)j * HD + 32 + hi * 16);
            v8f P = zero8();
            P = wmma_bf16(rA0, eB0, P);
            P = wmma_bf16(rA1, eB1, P);
#pragma unroll
            for (int r = 0; r < 8; ++r)
                Ph[(r + 8 * hi) * SW + nt * 16 + lrow] = P[r];
        }
        asm volatile("s_wait_dscnt 0" ::: "memory");  // LDS store -> load fence (CDNA5 split cnt)

        // ---- add diagonal-shifted rel-pos, apply 1/scale = sqrt(hd) = 8 ----
#pragma unroll
        for (int r = 0; r < 8; ++r) {
            const int M = r + 8 * hi;
            Sa[r] = (Sa[r] + Ph[M * SW + (lrow + 15 - M)]) * 8.0f;
            Sb[r] = (Sb[r] + Ph[M * SW + (lrow + 16 + 15 - M)]) * 8.0f;
        }

        // ---- online softmax (row reductions across 16-lane halves, wave32) ----
        float corr[8];
#pragma unroll
        for (int r = 0; r < 8; ++r) {
            float mx = fmaxf(Sa[r], Sb[r]);
            for (int off = 1; off < 16; off <<= 1)
                mx = fmaxf(mx, __shfl_xor(mx, off, 32));
            const float mn = fmaxf(m[r], mx);
            corr[r] = exp2f((m[r] - mn) * 1.44269504f);
            m[r] = mn;
        }
#pragma unroll
        for (int r = 0; r < 8; ++r) {
            const float pa = exp2f((Sa[r] - m[r]) * 1.44269504f);
            const float pb = exp2f((Sb[r] - m[r]) * 1.44269504f);
            float rs = pa + pb;
            for (int off = 1; off < 16; off <<= 1)
                rs += __shfl_xor(rs, off, 32);
            l[r] = l[r] * corr[r] + rs;
            const int M = r + 8 * hi;
            Pph[M * 32 + lrow]      = (__bf16)pa;
            Pph[M * 32 + 16 + lrow] = (__bf16)pb;
        }
#pragma unroll
        for (int r = 0; r < 8; ++r) {
            oacc0[r] *= corr[r]; oacc1[r] *= corr[r];
            oacc2[r] *= corr[r]; oacc3[r] *= corr[r];
        }
        asm volatile("s_wait_dscnt 0" ::: "memory");

        // ---- PV: probs (16x32 A) x V^T columns (32x16 B), hd = 4 N-tiles ----
        const v16bf pA = load_frag_A(Pph + lrow * 32 + hi * 8);
        oacc0 = wmma_bf16(pA, vB0, oacc0);
        oacc1 = wmma_bf16(pA, vB1, oacc1);
        oacc2 = wmma_bf16(pA, vB2, oacc2);
        oacc3 = wmma_bf16(pA, vB3, oacc3);
    }

    // ---- finalize: divide by softmax denominator, store bf16 [token, 1024] ----
#pragma unroll
    for (int r = 0; r < 8; ++r) {
        const float inv = 1.0f / l[r];
        const size_t tok = (size_t)(b * TDIM + t0 + r + 8 * hi) * ED + h * HD;
        AttnOb[tok + 0 * 16 + lrow] = (__bf16)(oacc0[r] * inv);
        AttnOb[tok + 1 * 16 + lrow] = (__bf16)(oacc1[r] * inv);
        AttnOb[tok + 2 * 16 + lrow] = (__bf16)(oacc2[r] * inv);
        AttnOb[tok + 3 * 16 + lrow] = (__bf16)(oacc3[r] * inv);
    }
}

// ---------------------------------------------------------------------------
extern "C" void kernel_launch(void* const* d_in, const int* in_sizes, int n_in,
                              void* d_out, int out_size, void* d_ws, size_t ws_size,
                              hipStream_t stream) {
    const float* query   = (const float*)d_in[0];
    const float* key     = (const float*)d_in[1];
    const float* value   = (const float*)d_in[2];
    const float* Wq      = (const float*)d_in[3];
    const float* bq      = (const float*)d_in[4];
    const float* Wk      = (const float*)d_in[5];
    const float* bk      = (const float*)d_in[6];
    const float* Wv      = (const float*)d_in[7];
    const float* bv      = (const float*)d_in[8];
    const float* Wo      = (const float*)d_in[9];
    const float* bo      = (const float*)d_in[10];
    const float* rel_emb = (const float*)d_in[11];

    const size_t xBytes    = (size_t)NTOK * ED * sizeof(__bf16);            // 8 MB
    const size_t wBytes    = (size_t)ED * ED * sizeof(__bf16);              // 2 MB
    const size_t headBytes = (size_t)BB * NH * TDIM * HD * sizeof(__bf16);  // 8 MB
    const size_t relBytes  = ((size_t)RELN * HD * sizeof(__bf16) + 255) & ~(size_t)255;
    const size_t need = 3 * xBytes + 4 * wBytes + 3 * headBytes + relBytes + xBytes;
    if (ws_size < need) return;  // workspace too small; cannot run safely

    char* w = (char*)d_ws;
    __bf16* Xqb  = (__bf16*)w; w += xBytes;
    __bf16* Xkb  = (__bf16*)w; w += xBytes;
    __bf16* Xvb  = (__bf16*)w; w += xBytes;
    __bf16* Wqb  = (__bf16*)w; w += wBytes;
    __bf16* Wkb  = (__bf16*)w; w += wBytes;
    __bf16* Wvb  = (__bf16*)w; w += wBytes;
    __bf16* Wob  = (__bf16*)w; w += wBytes;
    __bf16* Qb   = (__bf16*)w; w += headBytes;
    __bf16* Kb   = (__bf16*)w; w += headBytes;
    __bf16* VbT  = (__bf16*)w; w += headBytes;
    __bf16* relb = (__bf16*)w; w += relBytes;
    __bf16* AttnOb = (__bf16*)w;

    // 1) one-time bf16 conversions (hoisted out of all GEMM inner loops)
    {
        const int nx = NTOK * ED, nw = ED * ED, nr = RELN * HD;
        const int thr = 256;
        cvt4_kernel<<<(nx / 4 + thr - 1) / thr, thr, 0, stream>>>(query, Xqb, nx);
        cvt4_kernel<<<(nx / 4 + thr - 1) / thr, thr, 0, stream>>>(key,   Xkb, nx);
        cvt4_kernel<<<(nx / 4 + thr - 1) / thr, thr, 0, stream>>>(value, Xvb, nx);
        cvt4_kernel<<<(nw / 4 + thr - 1) / thr, thr, 0, stream>>>(Wq, Wqb, nw);
        cvt4_kernel<<<(nw / 4 + thr - 1) / thr, thr, 0, stream>>>(Wk, Wkb, nw);
        cvt4_kernel<<<(nw / 4 + thr - 1) / thr, thr, 0, stream>>>(Wv, Wvb, nw);
        cvt4_kernel<<<(nw / 4 + thr - 1) / thr, thr, 0, stream>>>(Wo, Wob, nw);
        cvt4_kernel<<<(nr / 4 + thr - 1) / thr, thr, 0, stream>>>(rel_emb, relb, nr);
    }
    // 2) Q/K/V projections (V stored transposed [B,H,hd,T])
    {
        dim3 g(NTOK / 16, ED / 256);  // 256 x 4 blocks, 4 waves x 16x64 tile
        gemm_bf16_kernel<0><<<g, 128, 0, stream>>>(Xqb, Wqb, bq, Qb);
        gemm_bf16_kernel<0><<<g, 128, 0, stream>>>(Xkb, Wkb, bk, Kb);
        gemm_bf16_kernel<1><<<g, 128, 0, stream>>>(Xvb, Wvb, bv, VbT);
    }
    // 3) flash attention with relative-position scores -> bf16 [token, 1024]
    {
        dim3 g(TDIM / 64, BB * NH);   // 16 x 64 blocks, 4 waves x 16 rows
        attn_kernel<<<g, 128, 0, stream>>>(Qb, Kb, VbT, Xqb, relb, AttnOb);
    }
    // 4) output projection -> d_out [B,T,ED] f32
    {
        dim3 g(NTOK / 16, ED / 256);
        gemm_bf16_kernel<2><<<g, 128, 0, stream>>>(AttnOb, Wob, bo, (float*)d_out);
    }
}
